// RGCN_83167746719887
// MI455X (gfx1250) — compile-verified, matrix operand order checked by reading, not run on hardware
//
#include <hip/hip_runtime.h>
#include <hip/hip_bf16.h>
#include <stdint.h>

// ---------------------------------------------------------------------------
// Problem constants (match reference file)
// ---------------------------------------------------------------------------
#define N_ENT  50000
#define R_REL  16
#define B_BAS  16
#define DEMB   256
#define DHID   128
#define NEDGE  300000
#define NQ     100000
#define LN_EPS 1e-5f

#define USE_ASYNC_LDS 1

typedef __attribute__((ext_vector_type(16))) __bf16 v16bf;
typedef __attribute__((ext_vector_type(8)))  float  v8f;

// ---------------------------------------------------------------------------
// Helpers
// ---------------------------------------------------------------------------
__device__ __forceinline__ unsigned short f2bf(float f) {
    unsigned u = __builtin_bit_cast(unsigned, f);
    u += 0x7FFFu + ((u >> 16) & 1u);          // round-to-nearest-even
    return (unsigned short)(u >> 16);
}

union Frag16 { uint4 u[2]; v16bf v; };

// A-matrix fragment (16x32 bf16, row-major source, row held by lane&15).
// Lanes 0-15 hold K = {0..7, 16..23}, lanes 16-31 hold K = {8..15, 24..31}:
// two contiguous 16-byte chunks per lane.
__device__ __forceinline__ v16bf load_afrag(const unsigned short* __restrict__ rowp,
                                            int k0, int lane, bool valid) {
    Frag16 f;
    if (valid) {
        int kb = k0 + ((lane & 16) ? 8 : 0);
        f.u[0] = *(const uint4*)(rowp + kb);
        f.u[1] = *(const uint4*)(rowp + kb + 16);
    } else {
        f.u[0] = make_uint4(0u, 0u, 0u, 0u);
        f.u[1] = make_uint4(0u, 0u, 0u, 0u);
    }
    return f.v;
}

// B fragment: two 16-byte chunks, elements p[0..7] and p[16..23] (global or LDS source).
__device__ __forceinline__ v16bf load_bfrag(const unsigned short* __restrict__ p) {
    Frag16 f;
    f.u[0] = *(const uint4*)(p);
    f.u[1] = *(const uint4*)(p + 16);
    return f.v;
}

// Cooperative LDS fill via CDNA5 async global->LDS copies (ASYNCcnt path).
// bytes must be a multiple of blockDim.x*16 (true for all call sites).
__device__ __forceinline__ void fill_lds(char* smem, unsigned ldsOff,
                                         const void* __restrict__ gsrc,
                                         unsigned bytes) {
#if USE_ASYNC_LDS
    unsigned long long base = (unsigned long long)(uintptr_t)gsrc;
    unsigned stride = (unsigned)blockDim.x * 16u;
    for (unsigned off = (unsigned)threadIdx.x * 16u; off < bytes; off += stride) {
        unsigned dst = ldsOff + off;               // byte offset in wave's LDS
        unsigned long long src = base + off;
        asm volatile("global_load_async_to_lds_b128 %0, %1, off"
                     :: "v"(dst), "v"(src) : "memory");
    }
    asm volatile("s_wait_asynccnt 0x0" ::: "memory");
#else
    const uint4* g = (const uint4*)gsrc;
    uint4* d = (uint4*)(smem + ldsOff);
    unsigned n = bytes >> 4;
    for (unsigned i = threadIdx.x; i < n; i += blockDim.x) d[i] = g[i];
#endif
}

// ---------------------------------------------------------------------------
// Small utility kernels
// ---------------------------------------------------------------------------
__global__ void k_zero_i32(int* __restrict__ p, int n) {
    int i = blockIdx.x * 256 + threadIdx.x;
    if (i < n) p[i] = 0;
}

__global__ void k_cvt_bf16(const float* __restrict__ in,
                           unsigned short* __restrict__ out, int n) {
    int i = blockIdx.x * 256 + threadIdx.x;
    if (i < n) out[i] = f2bf(in[i]);
}

// Pack B operand (K x N row-major f32, or basis-combined) into per-lane WMMA
// fragment order: out[r][s][t][lane][16], s = K/32 step, t = N/16 tile.
__global__ void k_pack_b(const float* __restrict__ W,
                         const float* __restrict__ coeff,
                         const float* __restrict__ bases,
                         int Bn, int K, int N, unsigned short* __restrict__ out,
                         int total) {
    int id = blockIdx.x * 256 + threadIdx.x;
    if (id >= total) return;
    int e    = id & 15;
    int lane = (id >> 4) & 31;
    int T = N >> 4, S = K >> 5;
    int tmp = id >> 9;
    int t = tmp % T;
    int s = (tmp / T) % S;
    int r = tmp / (T * S);
    int base   = (lane & 16) ? 8 : 0;
    int klocal = (e < 8) ? (base + e) : (16 + base + (e - 8));
    int k = s * 32 + klocal;
    int n = t * 16 + (lane & 15);
    float val;
    if (coeff) {                    // weights[r] = sum_b coeff[r,b] * bases[b]
        val = 0.f;
        for (int b = 0; b < Bn; ++b)
            val += coeff[r * Bn + b] * bases[((size_t)b * K + k) * N + n];
    } else {
        val = W[(size_t)k * N + n];
    }
    out[id] = f2bf(val);
}

// ---------------------------------------------------------------------------
// Edge counting-sort by relation
// ---------------------------------------------------------------------------
__global__ void k_hist(const int* __restrict__ et, int* __restrict__ counts) {
    int e = blockIdx.x * 256 + threadIdx.x;
    if (e < NEDGE) atomicAdd(&counts[et[e] & 15], 1);
}

__global__ void k_scan(const int* __restrict__ counts, int* __restrict__ offsets) {
    if (threadIdx.x == 0) {
        int acc = 0;
        for (int r = 0; r < R_REL; ++r) { offsets[r] = acc; acc += counts[r]; }
        offsets[R_REL] = acc;
    }
}

__global__ void k_scatter(const int* __restrict__ et,
                          const int* __restrict__ offsets,
                          int* __restrict__ cursor, int* __restrict__ sorted) {
    int e = blockIdx.x * 256 + threadIdx.x;
    if (e < NEDGE) {
        int r = et[e] & 15;
        int p = atomicAdd(&cursor[r], 1);
        sorted[offsets[r] + p] = e;
    }
}

// ---------------------------------------------------------------------------
// Self-loop GEMM: Out = Xb @ B (direct store). B cached in LDS once per block;
// 4 waves grid-stride over 16-row tiles. B fragments of a k-step preloaded
// into 8 distinct register groups so the 8 WMMAs issue back-to-back.
// ---------------------------------------------------------------------------
__global__ void __launch_bounds__(128) k_gemm_self(
        const unsigned short* __restrict__ Xb, int K,
        const unsigned short* __restrict__ Bp, float* __restrict__ Out,
        int nTiles) {
    extern __shared__ char smem[];
    int lane = threadIdx.x & 31;
    int wid  = threadIdx.x >> 5;
    int S = K >> 5;
    unsigned bBytes = (unsigned)S * 8 * 32 * 16 * 2;
    fill_lds(smem, 0, Bp, bBytes);
    __syncthreads();
    const unsigned short* sB = (const unsigned short*)smem;

    int hi = (lane >> 4) & 1;
    int n0 = lane & 15;
    for (int tile = blockIdx.x * 4 + wid; tile < nTiles; tile += gridDim.x * 4) {
        int m0 = tile * 16;
        const unsigned short* rowp = Xb + (size_t)(m0 + n0) * K;
        v8f acc[8] = {};
        for (int s = 0; s < S; ++s) {
            v16bf a = load_afrag(rowp, s * 32, lane, true);
            const unsigned short* bp = sB + ((unsigned)(s * 8) * 32 + lane) * 16;
            v16bf bf[8];
            #pragma unroll
            for (int t = 0; t < 8; ++t) bf[t] = load_bfrag(bp + t * 512);
            #pragma unroll
            for (int t = 0; t < 8; ++t)
                acc[t] = __builtin_amdgcn_wmma_f32_16x16x32_bf16(
                    false, a, false, bf[t], (short)0, acc[t], false, false);
        }
        for (int t = 0; t < 8; ++t)
            for (int v = 0; v < 8; ++v)
                Out[(size_t)(m0 + v + hi * 8) * DHID + t * 16 + n0] = acc[t][v];
    }
}

// ---------------------------------------------------------------------------
// Per-edge relation GEMM: gather x[col], transform by W[rel] (LDS-cached),
// scatter-add to out[row]. Edges pre-sorted by relation; grid = (chunks, R).
// ---------------------------------------------------------------------------
__global__ void __launch_bounds__(128) k_gemm_edge(
        const unsigned short* __restrict__ Xb, int K,
        const unsigned short* __restrict__ Wp,
        const int* __restrict__ sorted, const int* __restrict__ offsets,
        const int* __restrict__ rowIdx, const int* __restrict__ colIdx,
        float* __restrict__ Out) {
    extern __shared__ char smem[];
    int lane = threadIdx.x & 31;
    int wid  = threadIdx.x >> 5;
    int r = blockIdx.y;
    int S = K >> 5;
    unsigned bBytes = (unsigned)S * 8 * 32 * 16 * 2;
    fill_lds(smem, 0, Wp + (size_t)r * (bBytes >> 1), bBytes);
    __syncthreads();
    const unsigned short* sB = (const unsigned short*)smem;

    int begin = offsets[r], end = offsets[r + 1];
    int nTiles = (end - begin + 15) >> 4;
    int hi = (lane >> 4) & 1;
    int n0 = lane & 15;

    for (int tile = blockIdx.x * 4 + wid; tile < nTiles; tile += gridDim.x * 4) {
        int start = begin + tile * 16;
        int cnt = min(16, end - start);
        int srcNode = (n0 < cnt) ? colIdx[sorted[start + n0]] : -1;
        bool valid = (srcNode >= 0);
        const unsigned short* rowp = valid ? (Xb + (size_t)srcNode * K) : Xb;

        v8f acc[8] = {};
        for (int s = 0; s < S; ++s) {
            v16bf a = load_afrag(rowp, s * 32, lane, valid);
            const unsigned short* bp = sB + ((unsigned)(s * 8) * 32 + lane) * 16;
            v16bf bf[8];
            #pragma unroll
            for (int t = 0; t < 8; ++t) bf[t] = load_bfrag(bp + t * 512);
            #pragma unroll
            for (int t = 0; t < 8; ++t)
                acc[t] = __builtin_amdgcn_wmma_f32_16x16x32_bf16(
                    false, a, false, bf[t], (short)0, acc[t], false, false);
        }
        for (int v = 0; v < 8; ++v) {
            int mm = v + hi * 8;
            if (mm < cnt) {
                int dst = rowIdx[sorted[start + mm]];
                for (int t = 0; t < 8; ++t)
                    atomicAdd(&Out[(size_t)dst * DHID + t * 16 + n0], acc[t][v]);
            }
        }
    }
}

// ---------------------------------------------------------------------------
// Fused ReLU + LayerNorm over rows of 128, emitting bf16 for the next GEMM.
// One wave per row; blockDim = (32, 4).
// ---------------------------------------------------------------------------
__global__ void k_relu_ln(const float* __restrict__ X,
                          const float* __restrict__ g, const float* __restrict__ bb,
                          unsigned short* __restrict__ Outb) {
    int lane = threadIdx.x;
    int row = blockIdx.x * blockDim.y + threadIdx.y;
    const float* xr = X + (size_t)row * DHID;
    float v[4], s1 = 0.f, s2 = 0.f;
    for (int i = 0; i < 4; ++i) {
        float x = fmaxf(xr[i * 32 + lane], 0.f);
        v[i] = x; s1 += x; s2 += x * x;
    }
    for (int off = 16; off > 0; off >>= 1) {
        s1 += __shfl_xor(s1, off, 32);
        s2 += __shfl_xor(s2, off, 32);
    }
    float mu  = s1 * (1.f / DHID);
    float var = s2 * (1.f / DHID) - mu * mu;
    float inv = rsqrtf(var + LN_EPS);
    unsigned short* outr = Outb + (size_t)row * DHID;
    for (int i = 0; i < 4; ++i) {
        int c = i * 32 + lane;
        outr[c] = f2bf((v[i] - mu) * inv * g[c] + bb[c]);
    }
}

// ---------------------------------------------------------------------------
// Link prediction: [head|rel|tail] (Q x 512) @ w1 -> +b1 -> ReLU -> dot w2 +b2
// w1 pack (128 KB) cached in LDS; 4 waves grid-stride over 16-query tiles.
// ---------------------------------------------------------------------------
__global__ void __launch_bounds__(128) k_link(
        const unsigned short* __restrict__ Xb,    // N x 128 bf16 node feats
        const unsigned short* __restrict__ Relb,  // R x 256 bf16
        const unsigned short* __restrict__ W1p,   // packed K=512, N=128
        const int* __restrict__ hid, const int* __restrict__ rid,
        const int* __restrict__ tid,
        const float* __restrict__ b1, const float* __restrict__ w2,
        const float* __restrict__ b2, float* __restrict__ out, int nTiles) {
    extern __shared__ char smem[];
    int lane = threadIdx.x & 31;
    int wid  = threadIdx.x >> 5;
    fill_lds(smem, 0, W1p, 16u * 8 * 32 * 16 * 2);   // 128 KB
    __syncthreads();
    const unsigned short* sB = (const unsigned short*)smem;

    int n0 = lane & 15;
    int hi = (lane >> 4) & 1;
    float bias = b2[0];

    for (int qt = blockIdx.x * 4 + wid; qt < nTiles; qt += gridDim.x * 4) {
        int q0 = qt * 16;
        const unsigned short* hp = Xb + (size_t)hid[q0 + n0] * DHID;
        const unsigned short* rp = Relb + (size_t)rid[q0 + n0] * DEMB;
        const unsigned short* tp = Xb + (size_t)tid[q0 + n0] * DHID;

        v8f acc[8] = {};
        for (int s = 0; s < 16; ++s) {        // K = 512 = 16 steps of 32
            const unsigned short* src; int k0;
            if (s < 4)       { src = hp; k0 = s * 32; }
            else if (s < 12) { src = rp; k0 = (s - 4) * 32; }
            else             { src = tp; k0 = (s - 12) * 32; }
            v16bf a = load_afrag(src, k0, lane, true);
            const unsigned short* bp = sB + ((unsigned)(s * 8) * 32 + lane) * 16;
            v16bf bf[8];
            #pragma unroll
            for (int t = 0; t < 8; ++t) bf[t] = load_bfrag(bp + t * 512);
            #pragma unroll
            for (int t = 0; t < 8; ++t)
                acc[t] = __builtin_amdgcn_wmma_f32_16x16x32_bf16(
                    false, a, false, bf[t], (short)0, acc[t], false, false);
        }
        float psum[8];
        for (int v = 0; v < 8; ++v) {
            float p = 0.f;
            for (int t = 0; t < 8; ++t) {
                int c = t * 16 + n0;
                p += fmaxf(acc[t][v] + b1[c], 0.f) * w2[c];
            }
            psum[v] = p;
        }
        for (int off = 1; off < 16; off <<= 1)
            for (int v = 0; v < 8; ++v)
                psum[v] += __shfl_xor(psum[v], off, 32);
        if (n0 == 0)
            for (int v = 0; v < 8; ++v)
                out[q0 + hi * 8 + v] = psum[v] + bias;
    }
}

// ---------------------------------------------------------------------------
// Host driver
// ---------------------------------------------------------------------------
extern "C" void kernel_launch(void* const* d_in, const int* in_sizes, int n_in,
                              void* d_out, int out_size, void* d_ws, size_t ws_size,
                              hipStream_t stream) {
    const int*   edge_index   = (const int*)d_in[0];
    const int*   row          = edge_index;            // (E,) targets
    const int*   col          = edge_index + NEDGE;    // (E,) sources
    const int*   edge_type    = (const int*)d_in[1];
    const int*   head_ids     = (const int*)d_in[2];
    const int*   rel_ids      = (const int*)d_in[3];
    const int*   tail_ids     = (const int*)d_in[4];
    const float* entity_emb   = (const float*)d_in[5];
    const float* relation_emb = (const float*)d_in[6];
    const float* bases0       = (const float*)d_in[7];
    const float* coeff0       = (const float*)d_in[8];
    const float* selfw0       = (const float*)d_in[9];
    const float* bases1       = (const float*)d_in[10];
    const float* coeff1       = (const float*)d_in[11];
    const float* selfw1       = (const float*)d_in[12];
    const float* ln0_g        = (const float*)d_in[13];
    const float* ln0_b        = (const float*)d_in[14];
    const float* ln1_g        = (const float*)d_in[15];
    const float* ln1_b        = (const float*)d_in[16];
    const float* w1           = (const float*)d_in[17];
    const float* b1           = (const float*)d_in[18];
    const float* w2           = (const float*)d_in[19];
    const float* b2           = (const float*)d_in[20];
    float*       out          = (float*)d_out;

    // Bump-allocate workspace (~67 MB total)
    char* p = (char*)d_ws;
    auto alloc = [&](size_t bytes) -> char* {
        char* q = p; p += (bytes + 255) & ~(size_t)255; return q;
    };
    unsigned short* xb0  = (unsigned short*)alloc((size_t)N_ENT * DEMB * 2);
    unsigned short* xb1  = (unsigned short*)alloc((size_t)N_ENT * DHID * 2);
    unsigned short* relb = (unsigned short*)alloc((size_t)R_REL * DEMB * 2);
    unsigned short* S0p  = (unsigned short*)alloc((size_t)1  * 8  * 8 * 512 * 2);
    unsigned short* W0p  = (unsigned short*)alloc((size_t)16 * 8  * 8 * 512 * 2);
    unsigned short* S1p  = (unsigned short*)alloc((size_t)1  * 4  * 8 * 512 * 2);
    unsigned short* W1p  = (unsigned short*)alloc((size_t)16 * 4  * 8 * 512 * 2);
    unsigned short* WLp  = (unsigned short*)alloc((size_t)1  * 16 * 8 * 512 * 2);
    float* acc0          = (float*)alloc((size_t)N_ENT * DHID * 4); // out0 & out1
    int*   sorted        = (int*)alloc((size_t)NEDGE * 4);
    int*   counts        = (int*)alloc(16 * 4);
    int*   cursor        = (int*)alloc(16 * 4);
    int*   offsets       = (int*)alloc(17 * 4);
    unsigned short* xb2  = xb0;   // xb0 dead after layer-0 edge GEMM

    // 1) counting-sort edges by relation
    k_zero_i32<<<1, 32, 0, stream>>>(counts, 16);
    k_zero_i32<<<1, 32, 0, stream>>>(cursor, 16);
    k_hist   <<<(NEDGE + 255) / 256, 256, 0, stream>>>(edge_type, counts);
    k_scan   <<<1, 32, 0, stream>>>(counts, offsets);
    k_scatter<<<(NEDGE + 255) / 256, 256, 0, stream>>>(edge_type, offsets, cursor, sorted);

    // 2) bf16 conversions
    k_cvt_bf16<<<(N_ENT * DEMB + 255) / 256, 256, 0, stream>>>(entity_emb, xb0, N_ENT * DEMB);
    k_cvt_bf16<<<(R_REL * DEMB + 255) / 256, 256, 0, stream>>>(relation_emb, relb, R_REL * DEMB);

    // 3) pack all B operands into WMMA fragment order
    k_pack_b<<<(32768  + 255) / 256, 256, 0, stream>>>(selfw0, nullptr, nullptr, 0, 256, 128, S0p, 32768);
    k_pack_b<<<(524288 + 255) / 256, 256, 0, stream>>>(nullptr, coeff0, bases0, B_BAS, 256, 128, W0p, 524288);
    k_pack_b<<<(16384  + 255) / 256, 256, 0, stream>>>(selfw1, nullptr, nullptr, 0, 128, 128, S1p, 16384);
    k_pack_b<<<(262144 + 255) / 256, 256, 0, stream>>>(nullptr, coeff1, bases1, B_BAS, 128, 128, W1p, 262144);
    k_pack_b<<<(65536  + 255) / 256, 256, 0, stream>>>(w1, nullptr, nullptr, 0, 512, 128, WLp, 65536);

    dim3 egrid(32, R_REL);          // 512 blocks x 4 waves over edge tiles
    const unsigned sh0 = 8u * 8 * 32 * 16 * 2;   // 64 KB (K=256 packs)
    const unsigned sh1 = 4u * 8 * 32 * 16 * 2;   // 32 KB (K=128 packs)
    const unsigned shL = 16u * 8 * 32 * 16 * 2;  // 128 KB (K=512 pack)

    // 4) layer 0: self GEMM (init) + per-edge relation GEMM (atomic add)
    k_gemm_self<<<128, 128, sh0, stream>>>(xb0, 256, S0p, acc0, N_ENT / 16);
    k_gemm_edge<<<egrid, 128, sh0, stream>>>(xb0, 256, W0p, sorted, offsets, row, col, acc0);
    k_relu_ln  <<<N_ENT / 4, dim3(32, 4), 0, stream>>>(acc0, ln0_g, ln0_b, xb1);

    // 5) layer 1 (reuses acc0 buffer)
    k_gemm_self<<<128, 128, sh1, stream>>>(xb1, 128, S1p, acc0, N_ENT / 16);
    k_gemm_edge<<<egrid, 128, sh1, stream>>>(xb1, 128, W1p, sorted, offsets, row, col, acc0);
    k_relu_ln  <<<N_ENT / 4, dim3(32, 4), 0, stream>>>(acc0, ln1_g, ln1_b, xb2);

    // 6) link prediction
    k_link<<<256, 128, shL, stream>>>(xb2, relb, WLp, head_ids, rel_ids, tail_ids,
                                      b1, w2, b2, out, NQ / 16);
}
